// FullyConnectedLeakyNetwork_39376260169753
// MI455X (gfx1250) — matrix operand order repaired and checked by reading, not compiled, not run on hardware
//
#include <hip/hip_runtime.h>
#include <hip/hip_bf16.h>

#define NN 4096
#define NSTEPS 128
#define NITER 8
#define THREADS 256
#define KSPLIT 4                          // waves cooperating on one row tile
#define K_PER_WAVE (NN / KSPLIT)          // 1024
#define ROW_TILES 2                       // 16-row tiles per block
#define ROWS_PER_BLOCK (ROW_TILES * 16)   // 32
#define BLOCKS (NN / ROWS_PER_BLOCK)      // 128

typedef __attribute__((ext_vector_type(16))) __bf16 v16bf;
typedef __attribute__((ext_vector_type(8)))  __bf16 v8bf;
typedef __attribute__((ext_vector_type(8)))  float  v8f;

union Frag16 { v16bf v; v8bf h[2]; };

// ---------------------------------------------------------------------------
// One-time fp32 -> bf16 conversion of the weight matrix (W stays L2-resident:
// 32 MB << 192 MB global L2, so after the first pass no step touches HBM).
// ---------------------------------------------------------------------------
__global__ void convert_w_kernel(const float4* __restrict__ in,
                                 __bf16* __restrict__ out) {
    size_t i = (size_t)blockIdx.x * blockDim.x + threadIdx.x; // one float4 each
    float4 f = in[i];
    size_t o = i * 4;
    out[o + 0] = (__bf16)f.x;
    out[o + 1] = (__bf16)f.y;
    out[o + 2] = (__bf16)f.z;
    out[o + 3] = (__bf16)f.w;
}

// ---------------------------------------------------------------------------
// Init: spk0 = (x[0] == 1), zero the grid-barrier counter.
// ---------------------------------------------------------------------------
__global__ void init_kernel(const float* __restrict__ x0,
                            __bf16* __restrict__ spk0,
                            unsigned* __restrict__ counter) {
    int i = blockIdx.x * blockDim.x + threadIdx.x;
    if (i < NN) spk0[i] = (__bf16)((x0[i] == 1.0f) ? 1.0f : 0.0f);
    if (i == 0) *counter = 0u;
}

// ---------------------------------------------------------------------------
// Persistent SNN kernel: all 1024 sequential steps in one launch.
// 128 blocks x 256 threads (8 wave32s). Each 16-row tile is computed by
// KSPLIT=4 waves (K=1024 each) via bf16 WMMA; partial sums are combined in
// LDS. Spikes are double-buffered in global memory with an atomic-counter
// grid barrier between steps.
// ---------------------------------------------------------------------------
__global__ void __launch_bounds__(THREADS)
snn_persistent_kernel(const __bf16* __restrict__ Wbf,
                      const float* __restrict__ x,      // [NSTEPS, NN]
                      const float* __restrict__ mem_in, // [NN]
                      __bf16* spk_buf0, __bf16* spk_buf1,
                      float* __restrict__ out,          // [NSTEPS*NN + NN]
                      const float* __restrict__ beta_p,
                      const float* __restrict__ thr_p,
                      unsigned* counter) {
    __shared__ __bf16 s_spk[NN];                         // 8 KB spike stage
    __shared__ float  s_acc[KSPLIT][ROWS_PER_BLOCK];     // K-split partials

    const int tid  = threadIdx.x;
    const int lane = tid & 31;
    const int wave = tid >> 5;
    const int hi   = lane >> 4;          // lane-half (K interleave select)
    const int m    = lane & 15;          // A-matrix row within tile
    const int row_tile = wave >> 2;      // 0..1 : which 16-row tile
    const int ks       = wave & 3;       // 0..3 : which K quarter
    const int blockRow = blockIdx.x * ROWS_PER_BLOCK;
    const int waveRow  = blockRow + row_tile * 16;
    const int kbase    = ks * K_PER_WAVE;

    const float beta = *beta_p;
    const float thr  = *thr_p;

    // W row segment this lane streams for the A fragment.
    const __bf16* arow = Wbf + (size_t)(waveRow + m) * NN + kbase;

    // Membrane potential lives in a register for the whole 1024-step run.
    float mreg = 0.0f;
    if (tid < ROWS_PER_BLOCK) mreg = mem_in[blockRow + tid];

    __bf16* bufs[2] = {spk_buf0, spk_buf1};
    unsigned step = 0;

    for (int it = 0; it < NITER; ++it) {
        const bool last_iter = (it == NITER - 1);
        for (int t = 0; t < NSTEPS; ++t) {
            const __bf16* spk_r = bufs[step & 1u];
            __bf16*       spk_w = bufs[(step + 1u) & 1u];

            // Stage the spike vector in LDS (16B chunks, 2 per thread).
            for (int i = tid; i < NN / 8; i += THREADS)
                ((v8bf*)s_spk)[i] = ((const v8bf*)spk_r)[i];
            __syncthreads();

            // ---- partial W @ spk for this wave's 16 rows x 1024 K ----
            v8f c = {};
            for (int k0 = 0; k0 < K_PER_WAVE; k0 += 32) {
                Frag16 a, b;
                // A 16x32 bf16: lanes 0-15 hold K {0..7,16..23},
                // lanes 16-31 hold K {8..15,24..31}.
                const v8bf* pa = (const v8bf*)(arow + k0 + hi * 8);
                a.h[0] = pa[0];   // +0  .. +7  elements
                a.h[1] = pa[2];   // +16 .. +23 elements
                // B 32x16 bf16 (all 16 columns = broadcast spikes):
                // lanes 0-15 hold K {0..7,16..23}, lanes 16-31 hold K {8..15,24..31}.
                const v8bf* pb = (const v8bf*)(s_spk + kbase + k0 + hi * 8);
                b.h[0] = pb[0];
                b.h[1] = pb[2];
                c = __builtin_amdgcn_wmma_f32_16x16x32_bf16(
                        false, a.v, false, b.v, (short)0, c, false, false);
            }

            // D layout: lane 0 holds M=0..7 in c[0..7]; lane 16 holds M=8..15.
            if ((lane & 15) == 0) {
                int base = row_tile * 16 + hi * 8;
#pragma unroll
                for (int r = 0; r < 8; ++r) s_acc[ks][base + r] = c[r];
            }
            __syncthreads();

            // ---- LIF membrane update (one thread per row) ----
            if (tid < ROWS_PER_BLOCK) {
                int row = blockRow + tid;
                float wspk = s_acc[0][tid] + s_acc[1][tid] +
                             s_acc[2][tid] + s_acc[3][tid];
                float inp   = x[t * NN + row] + wspk;
                float reset = (mreg > thr) ? 1.0f : 0.0f;  // from prev mem
                mreg = beta * mreg + inp - reset * thr;
                float s = (mreg > thr) ? 1.0f : 0.0f;
                spk_w[row] = (__bf16)s;
                if (last_iter) out[t * NN + row] = s;                 // trace
                if (last_iter && t == NSTEPS - 1)
                    out[NSTEPS * NN + row] = mreg;                    // mem_f
            }

            // ---- grid-wide barrier (co-resident blocks) ----
            __threadfence();
            __syncthreads();
            if (tid == 0) {
                atomicAdd(counter, 1u);
                unsigned target = (step + 1u) * (unsigned)gridDim.x;
                while (__hip_atomic_load(counter, __ATOMIC_ACQUIRE,
                                         __HIP_MEMORY_SCOPE_AGENT) < target) {
                    __builtin_amdgcn_s_sleep(1);
                }
                __threadfence();
            }
            __syncthreads();
            ++step;
        }
    }
}

// ---------------------------------------------------------------------------
// Launch. Inputs: x[128*4096], mem[4096], weights[4096*4096], beta, threshold,
// n_steps, iterations. Output: trace[128*4096] ++ mem_f[4096] (float32).
// Workspace: bf16 W (32 MB) + 2 spike buffers + barrier counter (~33.6 MB).
// ---------------------------------------------------------------------------
extern "C" void kernel_launch(void* const* d_in, const int* in_sizes, int n_in,
                              void* d_out, int out_size, void* d_ws, size_t ws_size,
                              hipStream_t stream) {
    const float* x      = (const float*)d_in[0];
    const float* mem_in = (const float*)d_in[1];
    const float* W      = (const float*)d_in[2];
    const float* beta_p = (const float*)d_in[3];
    const float* thr_p  = (const float*)d_in[4];
    float* out = (float*)d_out;

    char* ws = (char*)d_ws;
    __bf16*   Wbf     = (__bf16*)ws;                               // 32 MB
    __bf16*   spk0    = (__bf16*)(ws + (size_t)NN * NN * 2);       // 8 KB
    __bf16*   spk1    = (__bf16*)(ws + (size_t)NN * NN * 2 + NN * 2);
    unsigned* counter = (unsigned*)(ws + (size_t)NN * NN * 2 + 2 * NN * 2);

    // fp32 -> bf16 weights: 16.7M elems, one float4 per thread.
    convert_w_kernel<<<(NN * (size_t)NN / 4) / THREADS, THREADS, 0, stream>>>(
        (const float4*)W, Wbf);

    init_kernel<<<NN / THREADS, THREADS, 0, stream>>>(x, spk0, counter);

    snn_persistent_kernel<<<BLOCKS, THREADS, 0, stream>>>(
        Wbf, x, mem_in, spk0, spk1, out, beta_p, thr_p, counter);
}
